// MoE_53893249630587
// MI455X (gfx1250) — matrix operand order repaired and checked by reading, compiled, not run on hardware
//
#include <hip/hip_runtime.h>
#include <hip/hip_bf16.h>
#include <math.h>

#if defined(__has_include)
#if __has_include(<hip/amd_detail/amd_gfx1250_TDM.h>)
#define MOE_TDM_6ARG 1
#endif
#endif

// Problem constants (match reference)
#define NE     8
#define DM     1024
#define DF     3584
#define TT     4096   // BATCH * POS
#define TILE_M 32     // tokens per workgroup tile

typedef __attribute__((ext_vector_type(16))) __bf16         v16bf;
typedef __attribute__((ext_vector_type(8)))  float          v8f;
typedef __attribute__((ext_vector_type(8)))  unsigned short u16x8;
typedef __attribute__((ext_vector_type(4)))  unsigned int   uint32x4;
typedef __attribute__((ext_vector_type(8)))  int            int32x8;
typedef __attribute__((ext_vector_type(4)))  int            int32x4;

union ABU {
    struct { u16x8 lo, hi; } s;
    v16bf v;
};

static __device__ __forceinline__ unsigned short f2bf(float f) {
    union { float f; unsigned int u; } c;
    c.f = f;
    unsigned int u = c.u;
    unsigned int r = (u + 0x7FFFu + ((u >> 16) & 1u)) >> 16;   // RNE
    return (unsigned short)r;
}

// silu via hw exp + hw rcp (1-ulp, plenty for bf16 output)
static __device__ __forceinline__ float silu_fast(float v) {
    return v * __builtin_amdgcn_rcpf(1.0f + __expf(-v));
}

// ---------------------------------------------------------------------------
// 0) zero counters + output (we accumulate with atomics)
// ---------------------------------------------------------------------------
__global__ void moe_zero_kernel(float* out, int n, int* cnt) {
    int i = blockIdx.x * blockDim.x + threadIdx.x;
    if (i < NE) cnt[i] = 0;
    for (int j = i; j < n; j += gridDim.x * blockDim.x) out[j] = 0.0f;
}

// ---------------------------------------------------------------------------
// 1) fp32 -> bf16 conversion (streaming)
// ---------------------------------------------------------------------------
__global__ void moe_cvt_kernel(const float* __restrict__ src,
                               unsigned short* __restrict__ dst, int n) {
    int i = blockIdx.x * blockDim.x + threadIdx.x;
    for (int j = i; j < n; j += gridDim.x * blockDim.x) dst[j] = f2bf(src[j]);
}

// ---------------------------------------------------------------------------
// 2) gating: logits -> softmax -> top2 -> renorm -> per-expert token lists
// ---------------------------------------------------------------------------
__global__ __launch_bounds__(128)
void moe_gate_kernel(const float* __restrict__ x, const float* __restrict__ wg,
                     int* __restrict__ cnt, int* __restrict__ tokidx,
                     float* __restrict__ tokw) {
    __shared__ float swg[NE][DM];   // 32 KB
    const int tid = threadIdx.x;
    for (int i = tid; i < NE * DM; i += blockDim.x)
        swg[i / DM][i % DM] = wg[i];
    __syncthreads();

    const int t = blockIdx.x * blockDim.x + tid;
    if (t >= TT) return;

    const float* xr = x + (size_t)t * DM;
    float lg[NE];
#pragma unroll
    for (int e = 0; e < NE; ++e) lg[e] = 0.0f;
    for (int d = 0; d < DM; ++d) {
        float xv = xr[d];
#pragma unroll
        for (int e = 0; e < NE; ++e) lg[e] += xv * swg[e][d];
    }
    float m = lg[0];
#pragma unroll
    for (int e = 1; e < NE; ++e) m = fmaxf(m, lg[e]);
    float se = 0.0f;
    float wt[NE];
#pragma unroll
    for (int e = 0; e < NE; ++e) { wt[e] = expf(lg[e] - m); se += wt[e]; }
    float inv = 1.0f / se;
#pragma unroll
    for (int e = 0; e < NE; ++e) wt[e] *= inv;

    int   i0 = 0;     float w0 = wt[0];
#pragma unroll
    for (int e = 1; e < NE; ++e) if (wt[e] > w0) { w0 = wt[e]; i0 = e; }
    int   i1 = -1;    float w1 = -1.0f;
#pragma unroll
    for (int e = 0; e < NE; ++e) if (e != i0 && wt[e] > w1) { w1 = wt[e]; i1 = e; }
    float s = w0 + w1;
    w0 /= s; w1 /= s;

    int p0 = atomicAdd(&cnt[i0], 1);
    tokidx[i0 * TT + p0] = t;
    tokw  [i0 * TT + p0] = w0;
    int p1 = atomicAdd(&cnt[i1], 1);
    tokidx[i1 * TT + p1] = t;
    tokw  [i1 * TT + p1] = w1;
}

// ---------------------------------------------------------------------------
// 3) routed SwiGLU MLP: one workgroup per (expert, 32-token tile), 16 waves.
//    x tile gathered into LDS via TDM gather-mode (tensor_load_to_lds).
//    Stage 1 (per wave): h[32x16] = silu(x @ W1_sliceT) * (x @ W3_sliceT), K=1024
//    Stage 2 (per wave): out[32x64] += h_chunk[32x256] @ W2_sliceT
//    Combine: atomicAdd(out[token][d], gate_w * acc)
// ---------------------------------------------------------------------------
__global__ __launch_bounds__(512, 1)
void moe_mlp_kernel(const unsigned short* __restrict__ xb,
                    const unsigned short* __restrict__ w1b,
                    const unsigned short* __restrict__ w3b,
                    const unsigned short* __restrict__ w2b,
                    const int* __restrict__ cnt,
                    const int* __restrict__ tokidx,
                    const float* __restrict__ tokw,
                    float* __restrict__ out) {
    const int e    = blockIdx.y;
    const int tile = blockIdx.x;
    const int nTok = cnt[e];
    if (tile * TILE_M >= nTok) return;

    __shared__ unsigned short lx[TILE_M][DM];    // x tile, bf16, 64 KB
    __shared__ unsigned short lh[TILE_M][256];   // h chunk, bf16, 16 KB
    __shared__ float swt[TILE_M];
    __shared__ int   stk[TILE_M];

    const int tid  = threadIdx.x;
    const int lane = tid & 31;
    const int wv   = tid >> 5;                   // 16 waves

    if (tid < TILE_M) {
        int slot = tile * TILE_M + tid;
        int tok = 0; float w = 0.0f;
        if (slot < nTok) {
            tok = tokidx[e * TT + slot];
            w   = tokw  [e * TT + slot];
        }
        stk[tid] = tok;
        swt[tid] = w;
    }
    __syncthreads();

    // --- TDM gather: 2 descriptors, 16 token rows each (2B elems, row = 1024) ---
    if (wv == 0) {
#pragma unroll
        for (int h = 0; h < 2; ++h) {
            unsigned long long ga = (unsigned long long)(uintptr_t)xb;
            uint32x4 g0;
            g0[0] = 0x80000001u;                               // gather_mode=1, 16b idx, count=1
            g0[1] = (unsigned int)(uintptr_t)&lx[h * 16][0];   // LDS dest offset
            g0[2] = (unsigned int)(ga & 0xFFFFFFFFu);          // global_addr lo
            g0[3] = (unsigned int)((ga >> 32) & 0x1FFFFFFu) | (2u << 30); // hi | type=2
            int32x8 g1;
            g1[0] = 0x00010000;              // workgroup_mask=0, data_size=2B
            g1[1] = (int)(1024u << 16);      // tensor_dim0 = 1024
            g1[2] = (int)(4096u << 16);      // tensor_dim1 = 4096 (rows)
            g1[3] = (int)(1024u << 16);      // tile_dim0 = 1024
            g1[4] = 16;                      // tile_dim1 = #valid indices
            g1[5] = 1024;                    // tensor_dim0_stride
            g1[6] = 0;
            g1[7] = 0;
            int d[8];
#pragma unroll
            for (int j = 0; j < 8; ++j)
                d[j] = (stk[h * 16 + 2 * j] & 0xFFFF) | (stk[h * 16 + 2 * j + 1] << 16);
            int32x4 g2 = {d[0], d[1], d[2], d[3]};
            int32x4 g3 = {d[4], d[5], d[6], d[7]};
#ifdef MOE_TDM_6ARG
            int32x8 gz = {};
            __builtin_amdgcn_tensor_load_to_lds(g0, g1, g2, g3, gz, 0);
#else
            __builtin_amdgcn_tensor_load_to_lds(g0, g1, g2, g3, 0);
#endif
        }
        __builtin_amdgcn_s_wait_tensorcnt(0);
    }
    __syncthreads();

    const unsigned short* W1e = w1b + (size_t)e * DF * DM;
    const unsigned short* W3e = w3b + (size_t)e * DF * DM;
    const unsigned short* W2e = w2b + (size_t)e * DM * DF;

    // per-lane WMMA operand layout (wave32):
    // A (16x32 bf16): lane row = lane&15, 16 contiguous halves along K at kb
    const int kb  = (lane < 16) ? 0 : 16;
    const int m_r = lane & 15;
    // B (32x16 bf16): lane column = lane&15, 16 contiguous halves along K at bk
    const int bn  = lane & 15;
    const int bk  = (lane >> 4) << 4;

    const int d0 = wv * 64;                 // this wave's 64 output dims

    v8f accO[2][4] = {};                    // [Mtile][Ntile] -> 32 x 64

    for (int chunk = 0; chunk < DF / 256; ++chunk) {   // 14 chunks
        // ---- stage 1: h tile columns [c0, c0+16) over full K = 1024 ----
        const int c0 = chunk * 256 + wv * 16;
        v8f h1a[2] = {};
        v8f h3a[2] = {};
        const unsigned short* b1row = W1e + (size_t)(c0 + bn) * DM + bk;
        const unsigned short* b3row = W3e + (size_t)(c0 + bn) * DM + bk;
#pragma unroll 2
        for (int k0 = 0; k0 < DM; k0 += 32) {
            ABU A0, A1, B1, B3;
            A0.s.lo = *(const u16x8*)&lx[m_r][k0 + kb];
            A0.s.hi = *(const u16x8*)&lx[m_r][k0 + kb + 8];
            A1.s.lo = *(const u16x8*)&lx[m_r + 16][k0 + kb];
            A1.s.hi = *(const u16x8*)&lx[m_r + 16][k0 + kb + 8];
            const u16x8* p1 = (const u16x8*)(b1row + k0);
            const u16x8* p3 = (const u16x8*)(b3row + k0);
            __builtin_prefetch(b1row + k0 + 256, 0, 1);
            __builtin_prefetch(b3row + k0 + 256, 0, 1);
            B1.s.lo = p1[0]; B1.s.hi = p1[1];
            B3.s.lo = p3[0]; B3.s.hi = p3[1];
            h1a[0] = __builtin_amdgcn_wmma_f32_16x16x32_bf16(false, A0.v, false, B1.v,
                                                             (short)0, h1a[0], false, false);
            h1a[1] = __builtin_amdgcn_wmma_f32_16x16x32_bf16(false, A1.v, false, B1.v,
                                                             (short)0, h1a[1], false, false);
            h3a[0] = __builtin_amdgcn_wmma_f32_16x16x32_bf16(false, A0.v, false, B3.v,
                                                             (short)0, h3a[0], false, false);
            h3a[1] = __builtin_amdgcn_wmma_f32_16x16x32_bf16(false, A1.v, false, B3.v,
                                                             (short)0, h3a[1], false, false);
        }
        // silu(h1) * h3 -> LDS (C/D layout: elem r is (M = r + 8*(lane>=16), N = lane&15))
#pragma unroll
        for (int mt = 0; mt < 2; ++mt) {
#pragma unroll
            for (int r = 0; r < 8; ++r) {
                float hv = silu_fast(h1a[mt][r]) * h3a[mt][r];
                int mm = mt * 16 + r + ((lane >> 4) << 3);
                lh[mm][wv * 16 + (lane & 15)] = f2bf(hv);
            }
        }
        __syncthreads();

        // ---- stage 2: accO += h_chunk[32x256] @ W2_sliceT (B reused across 2 M tiles) ----
#pragma unroll
        for (int kk = 0; kk < 256; kk += 32) {
            ABU A0, A1;
            A0.s.lo = *(const u16x8*)&lh[m_r][kk + kb];
            A0.s.hi = *(const u16x8*)&lh[m_r][kk + kb + 8];
            A1.s.lo = *(const u16x8*)&lh[m_r + 16][kk + kb];
            A1.s.hi = *(const u16x8*)&lh[m_r + 16][kk + kb + 8];
#pragma unroll
            for (int nt = 0; nt < 4; ++nt) {
                const unsigned short* brow =
                    W2e + (size_t)(d0 + nt * 16 + bn) * DF + chunk * 256 + kk + bk;
                ABU B;
                B.s.lo = ((const u16x8*)brow)[0];
                B.s.hi = ((const u16x8*)brow)[1];
                accO[0][nt] = __builtin_amdgcn_wmma_f32_16x16x32_bf16(
                    false, A0.v, false, B.v, (short)0, accO[0][nt], false, false);
                accO[1][nt] = __builtin_amdgcn_wmma_f32_16x16x32_bf16(
                    false, A1.v, false, B.v, (short)0, accO[1][nt], false, false);
            }
        }
        __syncthreads();
    }

    // ---- weighted combine into out (each token receives 2 expert contributions) ----
#pragma unroll
    for (int mt = 0; mt < 2; ++mt) {
#pragma unroll
        for (int nt = 0; nt < 4; ++nt) {
#pragma unroll
            for (int r = 0; r < 8; ++r) {
                int mm  = mt * 16 + r + ((lane >> 4) << 3);
                int tok = stk[mm];
                float v = accO[mt][nt][r] * swt[mm];
                atomicAdd(&out[(size_t)tok * DM + d0 + nt * 16 + (lane & 15)], v);
            }
        }
    }
}

// ---------------------------------------------------------------------------
// launch
// ---------------------------------------------------------------------------
extern "C" void kernel_launch(void* const* d_in, const int* in_sizes, int n_in,
                              void* d_out, int out_size, void* d_ws, size_t ws_size,
                              hipStream_t stream) {
    (void)in_sizes; (void)n_in; (void)out_size; (void)ws_size;

    const float* x  = (const float*)d_in[0];
    const float* Wg = (const float*)d_in[1];
    const float* W1 = (const float*)d_in[2];
    const float* W3 = (const float*)d_in[3];
    const float* W2 = (const float*)d_in[4];
    float* out = (float*)d_out;

    // workspace layout
    char* ws = (char*)d_ws;
    unsigned short* xb  = (unsigned short*)ws; ws += (size_t)TT * DM * 2;
    unsigned short* w1b = (unsigned short*)ws; ws += (size_t)NE * DF * DM * 2;
    unsigned short* w3b = (unsigned short*)ws; ws += (size_t)NE * DF * DM * 2;
    unsigned short* w2b = (unsigned short*)ws; ws += (size_t)NE * DM * DF * 2;
    int*   cnt    = (int*)ws;   ws += 256;
    int*   tokidx = (int*)ws;   ws += (size_t)NE * TT * 4;
    float* tokw   = (float*)ws; ws += (size_t)NE * TT * 4;

    moe_zero_kernel<<<1024, 256, 0, stream>>>(out, TT * DM, cnt);

    moe_cvt_kernel<<<2048, 256, 0, stream>>>(x,  xb,  TT * DM);
    moe_cvt_kernel<<<4096, 256, 0, stream>>>(W1, w1b, NE * DF * DM);
    moe_cvt_kernel<<<4096, 256, 0, stream>>>(W3, w3b, NE * DF * DM);
    moe_cvt_kernel<<<4096, 256, 0, stream>>>(W2, w2b, NE * DM * DF);

    moe_gate_kernel<<<TT / 128, 128, 0, stream>>>(x, Wg, cnt, tokidx, tokw);

    dim3 grid(TT / TILE_M, NE);
    moe_mlp_kernel<<<grid, 512, 0, stream>>>(xb, w1b, w3b, w2b,
                                             cnt, tokidx, tokw, out);
}
